// MultiHeadAttention_59012850647568
// MI455X (gfx1250) — matrix operand order repaired and checked by reading, not compiled, not run on hardware
//
#include <hip/hip_runtime.h>

#define D_MODEL 1024
#define N_HEADS 16
#define DKH     64          // d_k per head
#define SEQ     2048
#define BATCH   2
#define M_TOT   (BATCH*SEQ) // 4096
#define N_QKV   (3*D_MODEL) // 3072

typedef __attribute__((ext_vector_type(16))) _Float16 v16h;
typedef __attribute__((ext_vector_type(8)))  _Float16 v8h;
typedef __attribute__((ext_vector_type(2)))  _Float16 h2;
typedef __attribute__((ext_vector_type(8)))  float    v8f;
typedef __attribute__((ext_vector_type(4)))  int      v4i_;

union Frag { v16h v; v8h h[2]; };

static __device__ inline v8f wmma16(v16h a, v16h b, v8f c) {
  // D = A(16x32 f16) * B(32x16 f16) + C(16x16 f32)
  return __builtin_amdgcn_wmma_f32_16x16x32_f16(
      false, a, false, b, (short)0, c, false, false);
}

#if defined(__gfx1250__) && \
    __has_builtin(__builtin_amdgcn_global_load_async_to_lds_b128) && \
    __has_builtin(__builtin_amdgcn_s_wait_asynccnt)
#define USE_ASYNC_COPY 1
// per clang-22 diagnostic: param0 is  int4-vector addrspace(1)*  (printed as
// "__vector_size__(4*sizeof(int)) int __device__ *"); LDS side is addrspace(3)
typedef __attribute__((address_space(1))) v4i_* gv4;
typedef __attribute__((address_space(3))) v4i_* lv4;
#define ASYNC_B128(gp, lp) \
  __builtin_amdgcn_global_load_async_to_lds_b128((gv4)(gp), (lv4)(lp), 0, 0)
#else
#define USE_ASYNC_COPY 0
#endif

// ---------------------------------------------------------------------------
// Kernel 1: fused QKV projection.  C[m,e] = sum_d X[m,d]*Wqkv[e,d] + b[e]
// Block tile 128(M) x 128(N), K-chunk 32.  8 waves as 4(M) x 2(N); each wave
// owns a 32x64 tile: 2 A-frags x 4 B-frags -> 8 WMMAs per K-step.
// Global loads for chunk k+1 are issued before the WMMAs of chunk k.
// Epilogue scatters into Q[B,H,S,64] (x0.125), K[B,H,S,64], Vt[B,H,64,S].
// ---------------------------------------------------------------------------
__global__ __launch_bounds__(256) void qkv_proj_kernel(
    const float* __restrict__ x, const float* __restrict__ Wqkv,
    const float* __restrict__ bqkv,
    _Float16* __restrict__ qws, _Float16* __restrict__ kws,
    _Float16* __restrict__ vws)
{
  __shared__ _Float16 Xs[128 * 32];
  __shared__ _Float16 Ws2[128 * 32];

  const int tid  = threadIdx.x;
  const int wave = tid >> 5;
  const int lane = tid & 31;
  const int hf   = lane >> 4;
  const int l16  = lane & 15;
  const int wm   = wave >> 1;    // 0..3
  const int wn   = wave & 1;     // 0..1
  const int m0 = blockIdx.x * 128;
  const int n0 = blockIdx.y * 128;

  float4 xr[4], wr[4];
#pragma unroll
  for (int i = 0; i < 4; ++i) {
    const int idx = tid * 16 + i * 4;
    const int r = idx >> 5, c = idx & 31;
    xr[i] = *(const float4*)&x[(size_t)(m0 + r) * D_MODEL + c];
    wr[i] = *(const float4*)&Wqkv[(size_t)(n0 + r) * D_MODEL + c];
  }

  v8f acc[8] = {};   // [mt*4 + t]

  for (int kc = 0; kc < D_MODEL / 32; ++kc) {
    __syncthreads();
#pragma unroll
    for (int i = 0; i < 4; ++i) {
      const int idx = tid * 16 + i * 4;
      const int r = idx >> 5, c = idx & 31;
      Xs[r * 32 + c + 0] = (_Float16)xr[i].x;
      Xs[r * 32 + c + 1] = (_Float16)xr[i].y;
      Xs[r * 32 + c + 2] = (_Float16)xr[i].z;
      Xs[r * 32 + c + 3] = (_Float16)xr[i].w;
      Ws2[r * 32 + c + 0] = (_Float16)wr[i].x;
      Ws2[r * 32 + c + 1] = (_Float16)wr[i].y;
      Ws2[r * 32 + c + 2] = (_Float16)wr[i].z;
      Ws2[r * 32 + c + 3] = (_Float16)wr[i].w;
    }
    __syncthreads();
    if (kc + 1 < D_MODEL / 32) {
      const int k0n = (kc + 1) * 32;
#pragma unroll
      for (int i = 0; i < 4; ++i) {
        const int idx = tid * 16 + i * 4;
        const int r = idx >> 5, c = idx & 31;
        xr[i] = *(const float4*)&x[(size_t)(m0 + r) * D_MODEL + k0n + c];
        wr[i] = *(const float4*)&Wqkv[(size_t)(n0 + r) * D_MODEL + k0n + c];
      }
    }
    Frag a[2];
#pragma unroll
    for (int mt = 0; mt < 2; ++mt) {
      a[mt].h[0] = *(const v8h*)&Xs[(wm * 32 + mt * 16 + l16) * 32 + hf * 16];
      a[mt].h[1] = *(const v8h*)&Xs[(wm * 32 + mt * 16 + l16) * 32 + hf * 16 + 8];
    }
    v16h b[4];
#pragma unroll
    for (int t = 0; t < 4; ++t)
      b[t] = *(const v16h*)&Ws2[(wn * 64 + t * 16 + l16) * 32 + hf * 16];
#pragma unroll
    for (int mt = 0; mt < 2; ++mt)
#pragma unroll
      for (int t = 0; t < 4; ++t)
        acc[mt * 4 + t] = wmma16(a[mt].v, b[t], acc[mt * 4 + t]);
  }

  // epilogue: e = h*192 + part*64 + d  (part: 0=Q 1=K 2=V)
#pragma unroll
  for (int t = 0; t < 4; ++t) {
    const int e    = n0 + wn * 64 + t * 16 + l16;
    const int hd   = e / 192;
    const int rem  = e % 192;
    const int part = rem >> 6;
    const int d    = rem & 63;
    const float bias = bqkv[e];
#pragma unroll
    for (int mt = 0; mt < 2; ++mt) {
      const int mbase = m0 + wm * 32 + mt * 16 + hf * 8;
      if (part == 2) {
        // Vt[B,H,64,S]: consecutive r -> consecutive s, pack pairs
        const size_t vrow = ((size_t)((mbase >> 11) * N_HEADS + hd) * DKH + d) * SEQ;
#pragma unroll
        for (int r = 0; r < 8; r += 2) {
          const int s = (mbase + r) & (SEQ - 1);
          h2 pv;
          pv.x = (_Float16)(acc[mt * 4 + t][r] + bias);
          pv.y = (_Float16)(acc[mt * 4 + t][r + 1] + bias);
          *(h2*)&vws[vrow + s] = pv;
        }
      } else {
#pragma unroll
        for (int r = 0; r < 8; ++r) {
          const int m  = mbase + r;
          const int bb = m >> 11;
          const int s  = m & (SEQ - 1);
          const float val = acc[mt * 4 + t][r] + bias;
          if (part == 0)
            qws[((size_t)(bb * N_HEADS + hd) * SEQ + s) * DKH + d] = (_Float16)(val * 0.125f);
          else
            kws[((size_t)(bb * N_HEADS + hd) * SEQ + s) * DKH + d] = (_Float16)val;
        }
      }
    }
  }
}

// ---------------------------------------------------------------------------
// Kernel 2: flash-style attention.  One wave handles 32 query rows (2 tiles)
// across all keys; K and V fragments are loaded once per 32-key chunk and
// shared by both query tiles: 16 WMMAs per chunk from 8 B-frag loads.
// ---------------------------------------------------------------------------
__global__ __launch_bounds__(32) void attention_kernel(
    const _Float16* __restrict__ qws, const _Float16* __restrict__ kws,
    const _Float16* __restrict__ vws, _Float16* __restrict__ attn)
{
  __shared__ _Float16 Pstage[2][16 * 32];

  const int lane = threadIdx.x & 31;
  const int hf   = lane >> 4;
  const int l16  = lane & 15;
  const int bb = blockIdx.z, hd = blockIdx.y;
  const int q0 = blockIdx.x * 32;

  const _Float16* qbase = qws + (size_t)(bb * N_HEADS + hd) * SEQ * DKH;
  const _Float16* kbase = kws + (size_t)(bb * N_HEADS + hd) * SEQ * DKH;
  const _Float16* vbase = vws + (size_t)(bb * N_HEADS + hd) * DKH * SEQ;

  Frag aq[2][2];   // [qt][kk], resident for the whole key loop
#pragma unroll
  for (int qt = 0; qt < 2; ++qt) {
    const int qrow = q0 + qt * 16 + l16;
#pragma unroll
    for (int kk = 0; kk < 2; ++kk) {
      aq[qt][kk].h[0] = *(const v8h*)&qbase[qrow * DKH + kk * 32 + hf * 16];
      aq[qt][kk].h[1] = *(const v8h*)&qbase[qrow * DKH + kk * 32 + hf * 16 + 8];
    }
  }

  float mrow[2][8], lrow[2][8];
#pragma unroll
  for (int qt = 0; qt < 2; ++qt)
#pragma unroll
    for (int r = 0; r < 8; ++r) { mrow[qt][r] = -3.0e38f; lrow[qt][r] = 0.0f; }
  v8f o[2][4] = {};

  for (int kb = 0; kb < SEQ; kb += 32) {
    v16h bk[2][2];   // [st][kk], shared by both query tiles
#pragma unroll
    for (int st = 0; st < 2; ++st)
#pragma unroll
      for (int kk = 0; kk < 2; ++kk)
        bk[st][kk] = *(const v16h*)&kbase[(kb + st * 16 + l16) * DKH + kk * 32 + hf * 16];

    if (kb + 32 < SEQ) {   // prefetch next chunk (global_prefetch_b8)
      __builtin_prefetch(&kbase[(kb + 32 + l16) * DKH], 0, 0);
      __builtin_prefetch(&vbase[(size_t)(l16)*SEQ + kb + 32], 0, 0);
    }

    v8f s[2][2] = {};   // [qt][st]
#pragma unroll
    for (int qt = 0; qt < 2; ++qt)
#pragma unroll
      for (int kk = 0; kk < 2; ++kk)
#pragma unroll
        for (int st = 0; st < 2; ++st)
          s[qt][st] = wmma16(aq[qt][kk].v, bk[st][kk], s[qt][st]);

    // online softmax per query tile (C-layout rows live in one 16-lane half)
#pragma unroll
    for (int qt = 0; qt < 2; ++qt) {
      float p0[8], p1[8], alpha[8];
#pragma unroll
      for (int r = 0; r < 8; ++r) {
        float cm = fmaxf(s[qt][0][r], s[qt][1][r]);
#pragma unroll
        for (int off = 1; off < 16; off <<= 1)
          cm = fmaxf(cm, __shfl_xor(cm, off, 32));
        const float mn = fmaxf(mrow[qt][r], cm);
        alpha[r] = __expf(mrow[qt][r] - mn);
        mrow[qt][r] = mn;
        p0[r] = __expf(s[qt][0][r] - mn);
        p1[r] = __expf(s[qt][1][r] - mn);
        float sum = p0[r] + p1[r];
#pragma unroll
        for (int off = 1; off < 16; off <<= 1)
          sum += __shfl_xor(sum, off, 32);
        lrow[qt][r] = lrow[qt][r] * alpha[r] + sum;
      }
#pragma unroll
      for (int t = 0; t < 4; ++t)
#pragma unroll
        for (int r = 0; r < 8; ++r)
          o[qt][t][r] *= alpha[r];
#pragma unroll
      for (int r = 0; r < 8; ++r) {
        Pstage[qt][(hf * 8 + r) * 32 + l16]      = (_Float16)p0[r];
        Pstage[qt][(hf * 8 + r) * 32 + 16 + l16] = (_Float16)p1[r];
      }
    }
    __syncthreads();   // single-wave workgroup: cheap; keeps ordering explicit

    Frag pa[2];
#pragma unroll
    for (int qt = 0; qt < 2; ++qt) {
      pa[qt].h[0] = *(const v8h*)&Pstage[qt][l16 * 32 + hf * 16];
      pa[qt].h[1] = *(const v8h*)&Pstage[qt][l16 * 32 + hf * 16 + 8];
    }
    v16h bv[4];
#pragma unroll
    for (int t = 0; t < 4; ++t)
      bv[t] = *(const v16h*)&vbase[(size_t)(t * 16 + l16) * SEQ + kb + hf * 16];
#pragma unroll
    for (int qt = 0; qt < 2; ++qt)
#pragma unroll
      for (int t = 0; t < 4; ++t)
        o[qt][t] = wmma16(pa[qt].v, bv[t], o[qt][t]);
    __syncthreads();
  }

#pragma unroll
  for (int qt = 0; qt < 2; ++qt)
#pragma unroll
    for (int t = 0; t < 4; ++t)
#pragma unroll
      for (int r = 0; r < 8; ++r) {
        const int row = q0 + qt * 16 + hf * 8 + r;
        const float val = o[qt][t][r] / lrow[qt][r];
        attn[((size_t)(bb * SEQ + row)) * D_MODEL + hd * DKH + t * 16 + l16] = (_Float16)val;
      }
}

// ---------------------------------------------------------------------------
// Kernel 3: output projection.  out[m,e] = sum_d attn[m,d]*Wo[e,d] + bo[e]
// Same 128x128 block / 32x64 wave tiling.  The A tile is already f16, so on
// gfx1250 it is DMA'd straight into a double-buffered LDS tile with
// GLOBAL_LOAD_ASYNC_TO_LDS_B128 (ASYNCcnt), overlapping the WMMAs of the
// previous chunk; W goes through the fp32->f16 register pipeline.
// ---------------------------------------------------------------------------
__global__ __launch_bounds__(256) void out_proj_kernel(
    const _Float16* __restrict__ attn, const float* __restrict__ Wo,
    const float* __restrict__ bo, float* __restrict__ out)
{
#if USE_ASYNC_COPY
  __shared__ _Float16 As[2][128 * 32];
#else
  __shared__ _Float16 As[1][128 * 32];
#endif
  __shared__ _Float16 Ws2[128 * 32];

  const int tid  = threadIdx.x;
  const int wave = tid >> 5;
  const int lane = tid & 31;
  const int hf   = lane >> 4;
  const int l16  = lane & 15;
  const int wm   = wave >> 1;
  const int wn   = wave & 1;
  const int m0 = blockIdx.x * 128;
  const int n0 = blockIdx.y * 128;

  const int aidx = tid * 16;
  const int arow = aidx >> 5, acol = aidx & 31;
  const _Float16* gA = &attn[(size_t)(m0 + arow) * D_MODEL + acol];

  float4 wr[4];
#pragma unroll
  for (int i = 0; i < 4; ++i) {
    const int idx = tid * 16 + i * 4;
    const int r = idx >> 5, c = idx & 31;
    wr[i] = *(const float4*)&Wo[(size_t)(n0 + r) * D_MODEL + c];
  }

#if USE_ASYNC_COPY
  ASYNC_B128(gA,     &As[0][arow * 32 + acol]);
  ASYNC_B128(gA + 8, &As[0][arow * 32 + acol + 8]);
#else
  v8h ar[2];
  ar[0] = *(const v8h*)gA;
  ar[1] = *(const v8h*)(gA + 8);
#endif

  v8f acc[8] = {};

  for (int kc = 0; kc < D_MODEL / 32; ++kc) {
#if USE_ASYNC_COPY
    const int buf = kc & 1;
#else
    const int buf = 0;
#endif
    __syncthreads();   // previous chunk's readers done (LDS reusable)
#pragma unroll
    for (int i = 0; i < 4; ++i) {
      const int idx = tid * 16 + i * 4;
      const int r = idx >> 5, c = idx & 31;
      Ws2[r * 32 + c + 0] = (_Float16)wr[i].x;
      Ws2[r * 32 + c + 1] = (_Float16)wr[i].y;
      Ws2[r * 32 + c + 2] = (_Float16)wr[i].z;
      Ws2[r * 32 + c + 3] = (_Float16)wr[i].w;
    }
#if !USE_ASYNC_COPY
    *(v8h*)&As[0][arow * 32 + acol]     = ar[0];
    *(v8h*)&As[0][arow * 32 + acol + 8] = ar[1];
#endif
    if (kc + 1 < D_MODEL / 32) {
      const int k0n = (kc + 1) * 32;
#if USE_ASYNC_COPY
      ASYNC_B128(gA + k0n,     &As[buf ^ 1][arow * 32 + acol]);
      ASYNC_B128(gA + k0n + 8, &As[buf ^ 1][arow * 32 + acol + 8]);
#else
      ar[0] = *(const v8h*)(gA + k0n);
      ar[1] = *(const v8h*)(gA + k0n + 8);
#endif
#pragma unroll
      for (int i = 0; i < 4; ++i) {
        const int idx = tid * 16 + i * 4;
        const int r = idx >> 5, c = idx & 31;
        wr[i] = *(const float4*)&Wo[(size_t)(n0 + r) * D_MODEL + k0n + c];
      }
    }
#if USE_ASYNC_COPY
    if (kc + 1 < D_MODEL / 32) __builtin_amdgcn_s_wait_asynccnt(2);  // drain chunk kc
    else                       __builtin_amdgcn_s_wait_asynccnt(0);
#endif
    __syncthreads();   // all waves' A-tile DMA + W stores visible

    Frag a[2];
#pragma unroll
    for (int mt = 0; mt < 2; ++mt) {
      a[mt].h[0] = *(const v8h*)&As[buf][(wm * 32 + mt * 16 + l16) * 32 + hf * 16];
      a[mt].h[1] = *(const v8h*)&As[buf][(wm * 32 + mt * 16 + l16) * 32 + hf * 16 + 8];
    }
    v16h b[4];
#pragma unroll
    for (int t = 0; t < 4; ++t)
      b[t] = *(const v16h*)&Ws2[(wn * 64 + t * 16 + l16) * 32 + hf * 16];
#pragma unroll
    for (int mt = 0; mt < 2; ++mt)
#pragma unroll
      for (int t = 0; t < 4; ++t)
        acc[mt * 4 + t] = wmma16(a[mt].v, b[t], acc[mt * 4 + t]);
  }

#pragma unroll
  for (int t = 0; t < 4; ++t) {
    const int e = n0 + wn * 64 + t * 16 + l16;
    const float bias = bo[e];
#pragma unroll
    for (int mt = 0; mt < 2; ++mt)
#pragma unroll
      for (int r = 0; r < 8; ++r) {
        const int m = m0 + wm * 32 + mt * 16 + hf * 8 + r;
        out[(size_t)m * D_MODEL + e] = acc[mt * 4 + t][r] + bias;
      }
  }
}

// ---------------------------------------------------------------------------
extern "C" void kernel_launch(void* const* d_in, const int* in_sizes, int n_in,
                              void* d_out, int out_size, void* d_ws, size_t ws_size,
                              hipStream_t stream) {
  (void)in_sizes; (void)n_in; (void)out_size; (void)ws_size;
  const float* x    = (const float*)d_in[0];
  const float* Wqkv = (const float*)d_in[1];
  const float* bqkv = (const float*)d_in[2];
  const float* Wo   = (const float*)d_in[3];
  const float* bo   = (const float*)d_in[4];
  float* out = (float*)d_out;

  const size_t per = (size_t)BATCH * N_HEADS * SEQ * DKH; // 4M f16 = 8 MB
  _Float16* qws  = (_Float16*)d_ws;
  _Float16* kws  = qws + per;
  _Float16* vws  = kws + per;
  _Float16* attn = vws + per;   // total ws use: 32 MB

  qkv_proj_kernel<<<dim3(M_TOT / 128, N_QKV / 128), 256, 0, stream>>>(
      x, Wqkv, bqkv, qws, kws, vws);
  attention_kernel<<<dim3(SEQ / 32, N_HEADS, BATCH), 32, 0, stream>>>(
      qws, kws, vws, attn);
  out_proj_kernel<<<dim3(M_TOT / 128, D_MODEL / 128), 256, 0, stream>>>(
      attn, Wo, bo, out);
}
